// SimpleDecoderRetention_3161095930390
// MI455X (gfx1250) — matrix operand order repaired and checked by reading, not compiled, not run on hardware
//
#include <hip/hip_runtime.h>

// ---------------------------------------------------------------------------
// SimpleDecoderRetention on MI455X (gfx1250, wave32, WMMA 16x16x32 f16 + TDM)
//
//   K1: Q = xpos(X@W_Q), K = xpos(Mem@W_K, down) -> f16 [B][S][HD]
//       V = Mem@W_V -> f16 TRANSPOSED [B][HD][S] (so K2 can TDM it directly)
//   K2: fused chunkwise retention (64-wide key tiles), double-buffered TDM:
//       both K and V^T tiles fetched by TENSOR_LOAD_TO_LDS (async, TENSORcnt),
//       tile j+1 streams in while tile j is computed.
//       acc <- gamma^64 * acc + (QK^T .* gamma^(64-c)) @ V     (off-diag)
//       out <- gamma^r * acc + (QK^T .* causal gamma^(r-c)) @ V (diag)
//       Load balance: block x processes query tiles {x, 63-x} -> constant
//       65 key-tile iterations per block (perfect triangular balancing).
// ---------------------------------------------------------------------------

typedef __attribute__((ext_vector_type(16))) _Float16 v16h;
typedef __attribute__((ext_vector_type(8)))  float    v8f;
typedef __attribute__((ext_vector_type(4)))  unsigned v4u;
typedef __attribute__((ext_vector_type(8)))  int      v8i;
typedef __attribute__((ext_vector_type(4)))  int      v4i;

#define B_  4
#define S_  4096
#define H_  512
#define HD_ 128
#define NT_ (S_ / 64)             // 64 query/key tiles per sequence
#define LG2GAMMA (-0.04580369f)   // log2(0.96875)

__device__ __forceinline__ int koffA(int p, int kb) {
    return 2 * kb + 2 * p;
}

__device__ __forceinline__ v16h ldsB(const _Float16* p) {
    union { uint4 u[2]; v16h h; } t;
    const uint4* q = reinterpret_cast<const uint4*>(p);
    t.u[0] = q[0]; t.u[1] = q[1];
    return t.h;
}

// Tensor Data Mover: rows x cols f16 tile, global row stride `gstride` (elems),
// into LDS with pad_amount+1 DWORDs of pad every 2^(pad_interval+1) DWORDs.
__device__ __forceinline__ void tdm_load_tile_f16(unsigned lds_off,
                                                  const _Float16* gptr,
                                                  int rows, int cols, int gstride,
                                                  unsigned pad_interval,
                                                  unsigned pad_amount) {
    unsigned long long ga = (unsigned long long)(uintptr_t)gptr;
    v4u g0;
    g0[0] = 1u;                                       // count=1, user mode
    g0[1] = lds_off;                                  // lds_addr (bytes)
    g0[2] = (unsigned)(ga & 0xFFFFFFFFu);             // global_addr[31:0]
    g0[3] = (unsigned)((ga >> 32) & 0x01FFFFFFu)      // global_addr[56:32]
          | (2u << 30);                               // type=2 ("image")
    v8i g1;
    g1[0] = (int)((1u << 16)                          // data_size code 1 -> 2B
                | (1u << 20)                          // pad_enable
                | (pad_interval << 22)
                | (pad_amount << 25));
    g1[1] = (int)((unsigned)cols << 16);              // tensor_dim0[15:0]
    g1[2] = (int)((unsigned)rows << 16);              // dim0 hi=0 | tensor_dim1 lo
    g1[3] = (int)((unsigned)cols << 16);              // tensor_dim1 hi=0 | tile_dim0
    g1[4] = rows;                                     // tile_dim1 | tile_dim2=0
    g1[5] = gstride;                                  // tensor_dim0_stride[31:0]
    g1[6] = 0;
    g1[7] = 0;
    v4i z4 = {0, 0, 0, 0};
#if defined(__clang_major__) && (__clang_major__ >= 23)
    v8i z8 = {0, 0, 0, 0, 0, 0, 0, 0};
    __builtin_amdgcn_tensor_load_to_lds(g0, g1, z4, z4, z8, 0);
#else
    __builtin_amdgcn_tensor_load_to_lds(g0, g1, z4, z4, 0);
#endif
}

// ======================= Kernel 1: projection + xpos =======================
__global__ __launch_bounds__(128)
void proj_xpos_kernel(const float* __restrict__ X, const float* __restrict__ Mem,
                      const float* __restrict__ Wq, const float* __restrict__ Wk,
                      const float* __restrict__ Wv,
                      _Float16* __restrict__ Qo, _Float16* __restrict__ Ko,
                      _Float16* __restrict__ Vto) {
    __shared__ _Float16 Xl[64][72];
    __shared__ _Float16 Wt[128][72];   // W^T during GEMM; V^T staging afterwards

    const int tid  = threadIdx.x;
    const int lane = tid & 31, wave = tid >> 5;
    const int nl   = lane & 15;
    const int kb   = (lane < 16) ? 0 : 8;
    const int kb2  = (lane < 16) ? 0 : 16;
    const int hi8  = (lane < 16) ? 0 : 8;

    const int b = blockIdx.y, mode = blockIdx.z;   // 0:Q 1:K 2:V
    const int row0 = blockIdx.x * 64;
    const float* In = (mode == 0) ? X : Mem;
    const float* W  = (mode == 0) ? Wq : ((mode == 1) ? Wk : Wv);

    v8f acc[8] = {};

    for (int k0 = 0; k0 < H_; k0 += 64) {
        __syncthreads();
        float4 xr[8];
#pragma unroll
        for (int it = 0; it < 8; ++it) {
            int idx = tid + it * 128;
            int r = idx >> 4, c4 = (idx & 15) * 4;
            xr[it] = *reinterpret_cast<const float4*>(
                &In[(size_t)(b * S_ + row0 + r) * H_ + k0 + c4]);
        }
        float4 wr[16];
#pragma unroll
        for (int it = 0; it < 16; ++it) {
            int idx = tid + it * 128;
            int kk = idx >> 5, cb = (idx & 31) * 4;
            wr[it] = *reinterpret_cast<const float4*>(&W[(size_t)(k0 + kk) * HD_ + cb]);
        }
#pragma unroll
        for (int it = 0; it < 8; ++it) {
            int idx = tid + it * 128;
            int r = idx >> 4, c4 = (idx & 15) * 4;
            Xl[r][c4 + 0] = (_Float16)xr[it].x; Xl[r][c4 + 1] = (_Float16)xr[it].y;
            Xl[r][c4 + 2] = (_Float16)xr[it].z; Xl[r][c4 + 3] = (_Float16)xr[it].w;
        }
#pragma unroll
        for (int it = 0; it < 16; ++it) {
            int idx = tid + it * 128;
            int kk = idx >> 5, cb = (idx & 31) * 4;
            Wt[cb + 0][kk] = (_Float16)wr[it].x; Wt[cb + 1][kk] = (_Float16)wr[it].y;
            Wt[cb + 2][kk] = (_Float16)wr[it].z; Wt[cb + 3][kk] = (_Float16)wr[it].w;
        }
        __syncthreads();

        const int arow = wave * 16 + nl;
#pragma unroll
        for (int q = 0; q < 2; ++q) {
            union { unsigned int u[8]; v16h h; } A;
#pragma unroll
            for (int p = 0; p < 8; ++p)
                A.u[p] = *reinterpret_cast<const unsigned int*>(
                    &Xl[arow][q * 32 + koffA(p, kb)]);
            v16h bf[8];
#pragma unroll
            for (int nt = 0; nt < 8; ++nt)
                bf[nt] = ldsB(&Wt[nt * 16 + nl][q * 32 + kb2]);
#pragma unroll
            for (int nt = 0; nt < 8; ++nt)
                acc[nt] = __builtin_amdgcn_wmma_f32_16x16x32_f16(
                    false, A.h, false, bf[nt], (short)0, acc[nt], false, false);
        }
    }

    __syncthreads();   // Wt GEMM reads done; reusable as V^T staging

    if (mode == 2) {
        // stage V tile transposed in LDS, then coalesced store to Vt[B][HD][S]
#pragma unroll
        for (int nt = 0; nt < 8; ++nt)
#pragma unroll
            for (int v = 0; v < 8; ++v)
                Wt[nt * 16 + nl][wave * 16 + v + hi8] = (_Float16)acc[nt][v];
        __syncthreads();
        const int d = tid;   // 0..127
        const uint4* src = reinterpret_cast<const uint4*>(&Wt[d][0]);
        uint4* dst = reinterpret_cast<uint4*>(&Vto[((size_t)b * HD_ + d) * S_ + row0]);
#pragma unroll
        for (int t8 = 0; t8 < 8; ++t8) dst[t8] = src[t8];
        return;
    }

    // xpos rotary + scale; pair (2t,2t+1) lives on adjacent lanes -> shfl_xor(1)
    _Float16* Out = (mode == 0) ? Qo : Ko;
    const float lg2_10k_o64 = 0.2076217f;  // log2(10000)/64
#pragma unroll
    for (int nt = 0; nt < 8; ++nt) {
        int d = nt * 16 + nl;
        int t = d >> 1;
        float invf = exp2f(-lg2_10k_o64 * (float)t);
        float base = ((float)(2 * t) + 0.4f * 128.0f) / (1.4f * 128.0f);
        float lb   = log2f(base) * (1.0f / 512.0f);
#pragma unroll
        for (int v = 0; v < 8; ++v) {
            int   n = row0 + wave * 16 + v + hi8;
            float p = acc[nt][v];
            float ang = (float)n * invf;
            float sn = __sinf(ang), cs = __cosf(ang);
            float e  = lb * (float)n;
            float sc = exp2f((mode == 1) ? -e : e);
            float other = __shfl_xor(p, 1, 32);
            float val = ((lane & 1) == 0) ? sc * (p * cs - other * sn)
                                          : sc * (p * cs + other * sn);
            Out[(size_t)(b * S_ + n) * HD_ + d] = (_Float16)val;
        }
    }
}

// ======================= Kernel 2: fused retention =========================
__global__ __launch_bounds__(128)
void retention_kernel(const _Float16* __restrict__ Qg, const _Float16* __restrict__ Kg,
                      const _Float16* __restrict__ Vtg, float* __restrict__ Out) {
    __shared__ _Float16 Kl[2][64][136];   // K tiles (TDM, pad 4 DW -> stride 136)
    __shared__ _Float16 Vt[2][128][72];   // V^T tiles (TDM, pad 4 DW -> stride 72)
    __shared__ _Float16 Pst[4][16][72];   // per-wave C->A staging of P

    const int tid  = threadIdx.x;
    const int lane = tid & 31, wave = tid >> 5;
    const int nl   = lane & 15;
    const int kb   = (lane < 16) ? 0 : 8;
    const int kb2  = (lane < 16) ? 0 : 16;
    const int hi8  = (lane < 16) ? 0 : 8;

    const int b = blockIdx.y;
    const float g64 = exp2f(64.0f * LG2GAMMA);

    const unsigned kl_lds[2] = { (unsigned)(uintptr_t)&Kl[0][0][0],
                                 (unsigned)(uintptr_t)&Kl[1][0][0] };
    const unsigned vt_lds[2] = { (unsigned)(uintptr_t)&Vt[0][0][0],
                                 (unsigned)(uintptr_t)&Vt[1][0][0] };

    float w_off[4];
#pragma unroll
    for (int st = 0; st < 4; ++st)
        w_off[st] = exp2f((float)(64 - (st * 16 + nl)) * LG2GAMMA);

    // Triangular balancing: this block handles query tiles x and NT_-1-x,
    // total key-tile iterations = (x+1) + (NT_-x) = NT_+1, constant per block.
    for (int half = 0; half < 2; ++half) {
        const int i = half ? (NT_ - 1 - (int)blockIdx.x) : (int)blockIdx.x;

        union { unsigned int u[8]; v16h h; } AQ[4];
        const int qrow = i * 64 + wave * 16 + nl;
#pragma unroll
        for (int ch = 0; ch < 4; ++ch)
#pragma unroll
            for (int p = 0; p < 8; ++p)
                AQ[ch].u[p] = *reinterpret_cast<const unsigned int*>(
                    &Qg[(size_t)(b * S_ + qrow) * HD_ + ch * 32 + koffA(p, kb)]);

        v8f o[8] = {};

        for (int j = 0; j <= i; ++j) {
            const bool diag = (j == i);
            const int  buf  = j & 1;
            __syncthreads();   // all waves done reading buffer (j+1)&1

            if (wave == 0) {
                if (j == 0) {  // prime the pipeline: tile 0
                    tdm_load_tile_f16(kl_lds[0], &Kg[(size_t)(b * S_) * HD_],
                                      64, 128, 128, 5u, 3u);
                    tdm_load_tile_f16(vt_lds[0], &Vtg[(size_t)b * HD_ * S_],
                                      128, 64, S_, 4u, 3u);
                }
                if (j + 1 <= i) {  // stream tile j+1 into the other buffer
                    tdm_load_tile_f16(kl_lds[buf ^ 1],
                                      &Kg[(size_t)(b * S_ + (j + 1) * 64) * HD_],
                                      64, 128, 128, 5u, 3u);
                    tdm_load_tile_f16(vt_lds[buf ^ 1],
                                      &Vtg[(size_t)b * HD_ * S_ + (j + 1) * 64],
                                      128, 64, S_, 4u, 3u);
                    __builtin_amdgcn_s_wait_tensorcnt(2);  // tile j landed
                } else {
                    __builtin_amdgcn_s_wait_tensorcnt(0);
                }
            }
            __syncthreads();   // publish tile j to all waves

            // S = Q_i K_j^T  (16x64 per wave)
            v8f sacc[4] = {};
#pragma unroll
            for (int ch = 0; ch < 4; ++ch) {
                v16h bf[4];
#pragma unroll
                for (int st = 0; st < 4; ++st)
                    bf[st] = ldsB(&Kl[buf][st * 16 + nl][ch * 32 + kb2]);
#pragma unroll
                for (int st = 0; st < 4; ++st)
                    sacc[st] = __builtin_amdgcn_wmma_f32_16x16x32_f16(
                        false, AQ[ch].h, false, bf[st], (short)0, sacc[st],
                        false, false);
            }

            // decay weights + stage P as f16 for A-side of P@V
            if (!diag) {
#pragma unroll
                for (int nt = 0; nt < 8; ++nt) o[nt] = o[nt] * g64;
#pragma unroll
                for (int st = 0; st < 4; ++st)
#pragma unroll
                    for (int v = 0; v < 8; ++v)
                        Pst[wave][v + hi8][st * 16 + nl] =
                            (_Float16)(sacc[st][v] * w_off[st]);
            } else {
#pragma unroll
                for (int v = 0; v < 8; ++v) {
                    int r = wave * 16 + v + hi8;
                    float gr = exp2f((float)r * LG2GAMMA);
#pragma unroll
                    for (int nt = 0; nt < 8; ++nt) o[nt][v] *= gr;
                }
#pragma unroll
                for (int st = 0; st < 4; ++st) {
                    int c = st * 16 + nl;
#pragma unroll
                    for (int v = 0; v < 8; ++v) {
                        int r = wave * 16 + v + hi8;
                        float w = (r >= c) ? exp2f((float)(r - c) * LG2GAMMA) : 0.0f;
                        Pst[wave][v + hi8][c] = (_Float16)(sacc[st][v] * w);
                    }
                }
            }
            __syncthreads();

            // o += P @ V_j
#pragma unroll
            for (int q = 0; q < 2; ++q) {
                union { unsigned int u[8]; v16h h; } PA;
#pragma unroll
                for (int p = 0; p < 8; ++p)
                    PA.u[p] = *reinterpret_cast<const unsigned int*>(
                        &Pst[wave][nl][q * 32 + koffA(p, kb)]);
                v16h bv[8];
#pragma unroll
                for (int nt = 0; nt < 8; ++nt)
                    bv[nt] = ldsB(&Vt[buf][nt * 16 + nl][q * 32 + kb2]);
#pragma unroll
                for (int nt = 0; nt < 8; ++nt)
                    o[nt] = __builtin_amdgcn_wmma_f32_16x16x32_f16(
                        false, PA.h, false, bv[nt], (short)0, o[nt], false, false);
            }
        }

#pragma unroll
        for (int nt = 0; nt < 8; ++nt)
#pragma unroll
            for (int v = 0; v < 8; ++v) {
                int n = i * 64 + wave * 16 + v + hi8;
                Out[(size_t)(b * S_ + n) * HD_ + nt * 16 + nl] = o[nt][v];
            }
    }
}

// =============================== launcher ==================================
extern "C" void kernel_launch(void* const* d_in, const int* in_sizes, int n_in,
                              void* d_out, int out_size, void* d_ws, size_t ws_size,
                              hipStream_t stream) {
    const float* X   = (const float*)d_in[0];
    const float* Mem = (const float*)d_in[1];
    const float* Wq  = (const float*)d_in[2];
    const float* Wk  = (const float*)d_in[3];
    const float* Wv  = (const float*)d_in[4];
    float* out = (float*)d_out;

    _Float16* Qw  = (_Float16*)d_ws;                       // [B][S][HD]
    _Float16* Kw  = Qw + (size_t)B_ * S_ * HD_;            // [B][S][HD]
    _Float16* Vtw = Kw + (size_t)B_ * S_ * HD_;            // [B][HD][S]

    dim3 g1(S_ / 64, B_, 3), blk(128);
    proj_xpos_kernel<<<g1, blk, 0, stream>>>(X, Mem, Wq, Wk, Wv, Qw, Kw, Vtw);

    dim3 g2(NT_ / 2, B_);   // 32 x 4 perfectly balanced blocks
    retention_kernel<<<g2, blk, 0, stream>>>(Qw, Kw, Vtw, out);
}
